// MultiHeadPooledAttention_51324859187546
// MI455X (gfx1250) — compile-verified, matrix operand order checked by reading, not run on hardware
//
#include <hip/hip_runtime.h>
#include <hip/hip_bf16.h>
#include <math.h>

typedef __attribute__((ext_vector_type(16))) _Float16 v16h;
typedef __attribute__((ext_vector_type(8)))  _Float16 v8h;
typedef __attribute__((ext_vector_type(4)))  _Float16 v4h;
typedef __attribute__((ext_vector_type(8)))  float    v8f;
typedef __attribute__((ext_vector_type(4)))  float    v4f;

// Problem constants (fixed by setup_inputs: T=8,H=28,W=28,B=2,Dm=512,nh=8,hd=512)
static constexpr int TDIM  = 8;
static constexpr int HDIM  = 28;
static constexpr int WDIM  = 28;
static constexpr int BATCH = 2;
static constexpr int DM    = 512;
static constexpr int NHEAD = 8;
static constexpr int HD    = 512;
static constexpr int QDIM  = NHEAD * HD;          // 4096
static constexpr int LFULL = 1 + TDIM*HDIM*WDIM;  // 6273
static constexpr int H2    = 14;
static constexpr int W2    = 14;
static constexpr int BODY  = TDIM*H2*W2;          // 1568
static constexpr int LP    = 1 + BODY;            // 1569
static constexpr int KPOOL = 4 * HD;              // 2048

// Block tile: 128(M) x 128(N) x 32(K); 8 waves as 4(M) x 2(N);
// wave tile 32x64 = 2 A-frags x 4 B-frags = 8 wmma per K-step.
// Double-buffered LDS (prefetch next K-tile into regs while computing).

// ---------------------------------------------------------------------------
// LDS fragment loader matching CDNA5 16-bit A/B WMMA layout (wave32):
//   lanes 0-15 : row = lane,    elems 0..7 = K[0..7],  elems 8..15 = K[16..23]
//   lanes 16-31: row = lane-16, elems 0..7 = K[8..15], elems 8..15 = K[24..31]
__device__ __forceinline__ v16h ldsFrag(const _Float16* base, int lane) {
  int row = lane & 15;
  int kb  = (lane >> 4) << 3;           // 0 or 8
  const _Float16* p = base + row * 40 + kb;
  v8h lo = *(const v8h*)(p);
  v8h hi = *(const v8h*)(p + 16);
  return __builtin_shufflevector(lo, hi, 0,1,2,3,4,5,6,7,8,9,10,11,12,13,14,15);
}

// Epilogue: D layout -> lane<16: M=r, lane>=16: M=8+r ; N=lane&15
template<bool GUARD, int ACCC>
__device__ __forceinline__ void storeTile(float* __restrict__ C, int ldc,
                                          int M, int N,
                                          const float* __restrict__ bias,
                                          float alpha, int m0, int n0,
                                          int wm, int wn, int lane,
                                          v8f (&acc)[2][4])
{
  int colL = lane & 15;
  int rowH = (lane >> 4) * 8;
  #pragma unroll
  for (int j = 0; j < 4; ++j) {
    int col = n0 + wn + j * 16 + colL;
    if (GUARD && col >= N) continue;
    float bsum = bias ? bias[col] : 0.f;
    #pragma unroll
    for (int i = 0; i < 2; ++i) {
      #pragma unroll
      for (int r = 0; r < 8; ++r) {
        int row = m0 + wm + i * 16 + rowH + r;
        if (GUARD && row >= M) continue;
        long long idx = (long long)row * ldc + col;
        float v = alpha * acc[i][j][r] + bsum;
        if (ACCC) v += C[idx];
        C[idx] = v;
      }
    }
  }
}

// ---------------------------------------------------------------------------
// Generic batched GEMM: C = alpha * (A @ Bw^T) [+ bias] [+ C]
// A: [M,K] row-major (lda), batched by sA.
// VECA=1 requires lda%4==0 and K%32==0 (vectorized b128 staging).
// BTRANS=1: Bw is [N,K] row-major (ldb%4==0, K%32==0 required).
// BTRANS=0: Bw is [K,N] row-major (ldb%4==0, N%128==0 required; K arbitrary).
// C batch offset = (z/cDiv)*cOuter + (z%cDiv)*cInner ; C row stride = ldc.
template<int VECA, int BTRANS, int ACCC>
__global__ __launch_bounds__(256)
void wmma_gemm_kernel(const float* __restrict__ A, long long sA, int lda,
                      const float* __restrict__ Bw, long long sB, int ldb,
                      const float* __restrict__ bias,
                      float* __restrict__ C, long long cOuter, long long cInner,
                      int cDiv, int ldc,
                      int M, int N, int K, float alpha)
{
  __shared__ __align__(16) _Float16 lsA[2][128 * 40];
  __shared__ __align__(16) _Float16 lsB[2][128 * 40];

  int z = blockIdx.z;
  A  += (long long)z * sA;
  Bw += (long long)z * sB;
  C  += (long long)(z / cDiv) * cOuter + (long long)(z % cDiv) * cInner;

  int n0 = blockIdx.x * 128;
  int m0 = blockIdx.y * 128;
  int tid = threadIdx.x;
  int wave = tid >> 5, lane = tid & 31;
  int wm = (wave >> 1) * 32;   // 4 wave-rows of 32
  int wn = (wave & 1) * 64;    // 2 wave-cols of 64

  v8f acc[2][4];
  #pragma unroll
  for (int i = 0; i < 2; ++i)
    #pragma unroll
    for (int j = 0; j < 4; ++j)
      #pragma unroll
      for (int e = 0; e < 8; ++e) acc[i][j][e] = 0.f;

  // register staging for the prefetched K-tile
  v4f ra[4], rb[4];
  float ta[16];

  auto loadA = [&](int k0) {
    if (VECA) {
      #pragma unroll
      for (int it = 0; it < 4; ++it) {
        int q = tid + it * 256;
        int m = q >> 3, kv = (q & 7) * 4;
        int gm = m0 + m;
        bool ok = gm < M;
        ra[it] = *(const v4f*)(A + (long long)(ok ? gm : 0) * lda + (k0 + kv));
      }
    } else {
      #pragma unroll
      for (int it = 0; it < 16; ++it) {
        int q = tid + it * 256;
        int m = q >> 5, k = q & 31;
        int gm = m0 + m, gk = k0 + k;
        bool ok = (gm < M) && (gk < K);
        ta[it] = A[(long long)(ok ? gm : 0) * lda + (ok ? gk : 0)];
      }
    }
  };
  auto storeA = [&](int buf, int k0) {
    if (VECA) {
      #pragma unroll
      for (int it = 0; it < 4; ++it) {
        int q = tid + it * 256;
        int m = q >> 3, kv = (q & 7) * 4;
        bool ok = (m0 + m) < M;
        v4h h;
        #pragma unroll
        for (int e = 0; e < 4; ++e) h[e] = (_Float16)(ok ? ra[it][e] : 0.f);
        *(v4h*)&lsA[buf][m * 40 + kv] = h;
      }
    } else {
      #pragma unroll
      for (int it = 0; it < 16; ++it) {
        int q = tid + it * 256;
        int m = q >> 5, k = q & 31;
        bool ok = ((m0 + m) < M) && ((k0 + k) < K);
        lsA[buf][m * 40 + k] = (_Float16)(ok ? ta[it] : 0.f);
      }
    }
  };
  auto loadB = [&](int k0) {
    if (BTRANS) {
      #pragma unroll
      for (int it = 0; it < 4; ++it) {
        int q = tid + it * 256;
        int n = q >> 3, kv = (q & 7) * 4;
        int gn = n0 + n;
        bool ok = gn < N;
        rb[it] = *(const v4f*)(Bw + (long long)(ok ? gn : 0) * ldb + (k0 + kv));
      }
    } else {
      #pragma unroll
      for (int it = 0; it < 4; ++it) {
        int q = tid + it * 256;
        int k = q >> 5, n4 = (q & 31) * 4;
        int gk = k0 + k;
        bool ok = gk < K;
        rb[it] = *(const v4f*)(Bw + (long long)(ok ? gk : 0) * ldb + (n0 + n4));
      }
    }
  };
  auto storeB = [&](int buf, int k0) {
    if (BTRANS) {
      #pragma unroll
      for (int it = 0; it < 4; ++it) {
        int q = tid + it * 256;
        int n = q >> 3, kv = (q & 7) * 4;
        bool ok = (n0 + n) < N;
        v4h h;
        #pragma unroll
        for (int e = 0; e < 4; ++e) h[e] = (_Float16)(ok ? rb[it][e] : 0.f);
        *(v4h*)&lsB[buf][n * 40 + kv] = h;
      }
    } else {
      #pragma unroll
      for (int it = 0; it < 4; ++it) {
        int q = tid + it * 256;
        int k = q >> 5, n4 = (q & 31) * 4;
        bool ok = (k0 + k) < K;
        #pragma unroll
        for (int e = 0; e < 4; ++e)
          lsB[buf][(n4 + e) * 40 + k] = (_Float16)(ok ? rb[it][e] : 0.f);
      }
    }
  };
  auto compute = [&](int buf) {
    v16h af[2], bf[4];
    #pragma unroll
    for (int i = 0; i < 2; ++i) af[i] = ldsFrag(&lsA[buf][(wm + i * 16) * 40], lane);
    #pragma unroll
    for (int j = 0; j < 4; ++j) bf[j] = ldsFrag(&lsB[buf][(wn + j * 16) * 40], lane);
    #pragma unroll
    for (int i = 0; i < 2; ++i)
      #pragma unroll
      for (int j = 0; j < 4; ++j)
        acc[i][j] = __builtin_amdgcn_wmma_f32_16x16x32_f16(
            false, af[i], false, bf[j], (short)0, acc[i][j], false, false);
  };

  int nt = (K + 31) >> 5;
  loadA(0); loadB(0);
  storeA(0, 0); storeB(0, 0);
  __syncthreads();
  for (int t = 0; t < nt - 1; ++t) {
    int k1 = (t + 1) << 5;
    loadA(k1); loadB(k1);          // prefetch next tile (global -> regs)
    compute(t & 1);                 // compute current tile from LDS
    storeA((t + 1) & 1, k1);        // regs -> other LDS buffer
    storeB((t + 1) & 1, k1);
    __syncthreads();
  }
  compute((nt - 1) & 1);

  if (m0 + 128 <= M && n0 + 128 <= N)
    storeTile<false, ACCC>(C, ldc, M, N, bias, alpha, m0, n0, wm, wn, lane, acc);
  else
    storeTile<true,  ACCC>(C, ldc, M, N, bias, alpha, m0, n0, wm, wn, lane, acc);
}

// ---------------------------------------------------------------------------
// Pool conv (1,2,2)/stride(1,2,2) as gathered GEMM: M=BODY, N=512, K=2048.
// src row stride = srcRowStride (4096 for per-head QKV, 512 for x);
// batch z -> (b = z/nhInner, head n = z%nhInner). Writes body rows of dst
// (dst row 0 = cls handled separately). wr is repacked [c2][k4*512+ci].
__global__ __launch_bounds__(256)
void pool_gemm_kernel(const float* __restrict__ src, int srcRowStride, int nhInner,
                      const float* __restrict__ wr, float* __restrict__ dst)
{
  __shared__ __align__(16) _Float16 lsA[2][128 * 40];
  __shared__ __align__(16) _Float16 lsB[2][128 * 40];

  int z = blockIdx.z;
  int b = z / nhInner, n = z % nhInner;
  long long base = (long long)b * LFULL * srcRowStride + (long long)n * HD;
  float* C = dst + (long long)z * (LP * HD) + HD;   // skip cls row

  int n0 = blockIdx.x * 128;
  int m0 = blockIdx.y * 128;
  int tid = threadIdx.x;
  int wave = tid >> 5, lane = tid & 31;
  int wm = (wave >> 1) * 32, wn = (wave & 1) * 64;

  v8f acc[2][4];
  #pragma unroll
  for (int i = 0; i < 2; ++i)
    #pragma unroll
    for (int j = 0; j < 4; ++j)
      #pragma unroll
      for (int e = 0; e < 8; ++e) acc[i][j][e] = 0.f;

  v4f ra[4], rb[4];

  auto loadA = [&](int k0) {
    int k4 = k0 >> 9;                 // tap index, constant within K-tile
    int kh = k4 >> 1, kw = k4 & 1;
    int cibase = k0 & 511;
    #pragma unroll
    for (int it = 0; it < 4; ++it) {
      int q = tid + it * 256;
      int m = q >> 3, kv = (q & 7) * 4;
      int p = m0 + m;
      bool ok = p < BODY;
      int pc = ok ? p : 0;
      int t = pc / (H2 * W2);
      int r = pc % (H2 * W2);
      int hh = r / W2, ww = r % W2;
      int pos = t * (HDIM * WDIM) + (2 * hh + kh) * WDIM + (2 * ww + kw);
      ra[it] = *(const v4f*)(src + base + (long long)(1 + pos) * srcRowStride
                             + cibase + kv);
    }
  };
  auto storeA = [&](int buf) {
    #pragma unroll
    for (int it = 0; it < 4; ++it) {
      int q = tid + it * 256;
      int m = q >> 3, kv = (q & 7) * 4;
      bool ok = (m0 + m) < BODY;
      v4h h;
      #pragma unroll
      for (int e = 0; e < 4; ++e) h[e] = (_Float16)(ok ? ra[it][e] : 0.f);
      *(v4h*)&lsA[buf][m * 40 + kv] = h;
    }
  };
  auto loadB = [&](int k0) {
    #pragma unroll
    for (int it = 0; it < 4; ++it) {
      int q = tid + it * 256;
      int nn = q >> 3, kv = (q & 7) * 4;
      rb[it] = *(const v4f*)(wr + (long long)(n0 + nn) * KPOOL + k0 + kv);
    }
  };
  auto storeB = [&](int buf) {
    #pragma unroll
    for (int it = 0; it < 4; ++it) {
      int q = tid + it * 256;
      int nn = q >> 3, kv = (q & 7) * 4;
      v4h h;
      #pragma unroll
      for (int e = 0; e < 4; ++e) h[e] = (_Float16)rb[it][e];
      *(v4h*)&lsB[buf][nn * 40 + kv] = h;
    }
  };
  auto compute = [&](int buf) {
    v16h af[2], bf[4];
    #pragma unroll
    for (int i = 0; i < 2; ++i) af[i] = ldsFrag(&lsA[buf][(wm + i * 16) * 40], lane);
    #pragma unroll
    for (int j = 0; j < 4; ++j) bf[j] = ldsFrag(&lsB[buf][(wn + j * 16) * 40], lane);
    #pragma unroll
    for (int i = 0; i < 2; ++i)
      #pragma unroll
      for (int j = 0; j < 4; ++j)
        acc[i][j] = __builtin_amdgcn_wmma_f32_16x16x32_f16(
            false, af[i], false, bf[j], (short)0, acc[i][j], false, false);
  };

  constexpr int NT = KPOOL / 32;   // 64 K-tiles
  loadA(0); loadB(0);
  storeA(0); storeB(0);
  __syncthreads();
  for (int t = 0; t < NT - 1; ++t) {
    int k1 = (t + 1) << 5;
    loadA(k1); loadB(k1);
    compute(t & 1);
    storeA((t + 1) & 1);
    storeB((t + 1) & 1);
    __syncthreads();
  }
  compute((NT - 1) & 1);

  // N=512 is a multiple of 128 -> cols always in range; guard rows only.
  if (m0 + 128 <= BODY)
    storeTile<false, 0>(C, HD, BODY, HD, nullptr, 1.f, m0, n0, wm, wn, lane, acc);
  else
    storeTile<true,  0>(C, HD, BODY, HD, nullptr, 1.f, m0, n0, wm, wn, lane, acc);
}

// ---------------------------------------------------------------------------
__global__ void repack_w_kernel(const float* __restrict__ w, float* __restrict__ wr)
{
  int i = blockIdx.x * 256 + threadIdx.x;
  if (i >= HD * KPOOL) return;
  int c2 = i >> 11, c = i & 2047;
  int k4 = c >> 9, ci = c & 511;
  wr[i] = w[c2 * KPOOL + ci * 4 + k4];
}

__device__ __forceinline__ float sincos_val(int jj, int half, float pos)
{
  int idx = (jj < half) ? jj : jj - half;
  float omega = __expf(-9.210340371976184f * (float)idx / (float)half); // 10000^(-idx/half)
  float ang = pos * omega;
  return (jj < half) ? __sinf(ang) : __cosf(ang);
}

__global__ void emb_kernel(float* __restrict__ emb)
{
  int i = blockIdx.x * 256 + threadIdx.x;
  if (i >= BODY * HD) return;
  int p = i >> 9, j = i & 511;
  int t = p / (H2 * W2);
  int r = p % (H2 * W2);
  int h = r / W2, w = r % W2;
  float v;
  if (j < 170)      v = sincos_val(j,       85, (float)t);   // dt=170
  else if (j < 340) v = sincos_val(j - 170, 85, (float)h);   // dt=170
  else              v = sincos_val(j - 340, 86, (float)w);   // dw=172
  emb[i] = v;
}

__global__ void cls_copy_kernel(const float* __restrict__ src, int srcRowStride,
                                int nhInner, float* __restrict__ dst, int nbatch)
{
  int i = blockIdx.x * 256 + threadIdx.x;
  int z = i >> 9, d = i & 511;
  if (z >= nbatch) return;
  int b = z / nhInner, n = z % nhInner;
  dst[(long long)z * (LP * HD) + d] =
      src[(long long)b * LFULL * srcRowStride + (long long)n * HD + d];
}

__global__ __launch_bounds__(256)
void softmax_kernel(float* __restrict__ logits)
{
  long long row = blockIdx.x;
  float* p = logits + row * LP;
  __shared__ float red[8];
  int tid = threadIdx.x;

  float mx = -3.0e38f;
  for (int c = tid; c < LP; c += 256) mx = fmaxf(mx, p[c]);
  for (int o = 16; o > 0; o >>= 1) mx = fmaxf(mx, __shfl_xor(mx, o, 32));
  if ((tid & 31) == 0) red[tid >> 5] = mx;
  __syncthreads();
  float m = red[0];
  #pragma unroll
  for (int i = 1; i < 8; ++i) m = fmaxf(m, red[i]);
  __syncthreads();

  float s = 0.f;
  for (int c = tid; c < LP; c += 256) { float e = __expf(p[c] - m); p[c] = e; s += e; }
  for (int o = 16; o > 0; o >>= 1) s += __shfl_xor(s, o, 32);
  if ((tid & 31) == 0) red[tid >> 5] = s;
  __syncthreads();
  float tot = 0.f;
  #pragma unroll
  for (int i = 0; i < 8; ++i) tot += red[i];
  float inv = 1.f / tot;
  for (int c = tid; c < LP; c += 256) p[c] *= inv;
}

// stacked[b,q,n*HD+d] += (q==0 ? 1 : 2) * PQ[b,n,q,d]
__global__ void residual_kernel(float* __restrict__ stacked, const float* __restrict__ PQ)
{
  long long i = (long long)blockIdx.x * 256 + threadIdx.x;
  const long long total = (long long)BATCH * LP * QDIM;
  if (i >= total) return;
  int d = (int)(i & (QDIM - 1));
  long long t = i >> 12;
  int q = (int)(t % LP);
  int b = (int)(t / LP);
  int n = d >> 9, dd = d & 511;
  float f = (q == 0) ? 1.f : 2.f;
  stacked[i] += f * PQ[((long long)(b * NHEAD + n) * LP + q) * HD + dd];
}

// out = LN(y + px) * gamma + beta  (row = 512 wide, 256 threads x 2 elems)
__global__ __launch_bounds__(256)
void ln_kernel(const float* __restrict__ y, const float* __restrict__ px,
               const float* __restrict__ gamma, const float* __restrict__ beta,
               float* __restrict__ out)
{
  int row = blockIdx.x;
  const float* yp = y   + (long long)row * DM;
  const float* pp = px  + (long long)row * DM;
  float*       op = out + (long long)row * DM;
  int tid = threadIdx.x;
  float v0 = yp[tid]       + pp[tid];
  float v1 = yp[tid + 256] + pp[tid + 256];

  __shared__ float red[8];
  float s = v0 + v1;
  for (int o = 16; o > 0; o >>= 1) s += __shfl_xor(s, o, 32);
  if ((tid & 31) == 0) red[tid >> 5] = s;
  __syncthreads();
  float tot = 0.f;
  #pragma unroll
  for (int i = 0; i < 8; ++i) tot += red[i];
  float mu = tot * (1.f / 512.f);
  __syncthreads();

  float d0 = v0 - mu, d1 = v1 - mu;
  float sq = d0 * d0 + d1 * d1;
  for (int o = 16; o > 0; o >>= 1) sq += __shfl_xor(sq, o, 32);
  if ((tid & 31) == 0) red[tid >> 5] = sq;
  __syncthreads();
  float tv = 0.f;
  #pragma unroll
  for (int i = 0; i < 8; ++i) tv += red[i];
  float inv = rsqrtf(tv * (1.f / 512.f) + 1e-5f);

  op[tid]       = d0 * inv * gamma[tid]       + beta[tid];
  op[tid + 256] = d1 * inv * gamma[tid + 256] + beta[tid + 256];
}

// ---------------------------------------------------------------------------
extern "C" void kernel_launch(void* const* d_in, const int* in_sizes, int n_in,
                              void* d_out, int out_size, void* d_ws, size_t ws_size,
                              hipStream_t stream)
{
  const float* x     = (const float*)d_in[0];
  const float* Wq    = (const float*)d_in[1];
  const float* bq    = (const float*)d_in[2];
  const float* Wk    = (const float*)d_in[3];
  const float* bk    = (const float*)d_in[4];
  const float* Wv    = (const float*)d_in[5];
  const float* bv    = (const float*)d_in[6];
  const float* wpq   = (const float*)d_in[7];
  const float* wpk   = (const float*)d_in[8];
  const float* wpv   = (const float*)d_in[9];
  const float* wpx   = (const float*)d_in[10];
  const float* Wd    = (const float*)d_in[11];
  const float* bd    = (const float*)d_in[12];
  const float* gamma = (const float*)d_in[13];
  const float* beta  = (const float*)d_in[14];
  float* out = (float*)d_out;

  float* ws = (float*)d_ws;
  const long long BL = (long long)BATCH * LFULL;   // 12546
  long long off = 0;
  float* Q   = ws + off; off += BL * QDIM;
  float* Kb  = ws + off; off += BL * QDIM;
  float* V   = ws + off; off += BL * QDIM;
  float* PQ  = ws + off; off += (long long)BATCH * NHEAD * LP * HD;
  float* PK  = ws + off; off += (long long)BATCH * NHEAD * LP * HD;
  float* PV  = ws + off; off += (long long)BATCH * NHEAD * LP * HD;
  float* LG  = ws + off; off += (long long)BATCH * NHEAD * LP * LP;
  float* EM  = ws + off; off += (long long)BODY * HD;
  float* ST  = ws + off; off += (long long)BATCH * LP * QDIM;
  float* PX  = ws + off; off += (long long)BATCH * LP * HD;
  float* Y   = ws + off; off += (long long)BATCH * LP * DM;
  float* WRQ = ws + off; off += (long long)HD * KPOOL;
  float* WRK = ws + off; off += (long long)HD * KPOOL;
  float* WRV = ws + off; off += (long long)HD * KPOOL;
  float* WRX = ws + off; off += (long long)HD * KPOOL;
  (void)in_sizes; (void)n_in; (void)out_size; (void)ws_size;

  dim3 blk(256);

  // 1) repack pooling weights + positional table
  int rp = (HD * KPOOL + 255) / 256;
  repack_w_kernel<<<rp, blk, 0, stream>>>(wpq, WRQ);
  repack_w_kernel<<<rp, blk, 0, stream>>>(wpk, WRK);
  repack_w_kernel<<<rp, blk, 0, stream>>>(wpv, WRV);
  repack_w_kernel<<<rp, blk, 0, stream>>>(wpx, WRX);
  emb_kernel<<<(BODY * HD + 255) / 256, blk, 0, stream>>>(EM);

  // 2) Q/K/V projections: [12546,512] @ [4096,512]^T + bias
  {
    dim3 g(QDIM / 128, (int)((BL + 127) / 128), 1);
    wmma_gemm_kernel<1,1,0><<<g, blk, 0, stream>>>(x, 0, DM, Wq, 0, DM, bq,
                                                   Q, 0, 0, 1, QDIM, (int)BL, QDIM, DM, 1.f);
    wmma_gemm_kernel<1,1,0><<<g, blk, 0, stream>>>(x, 0, DM, Wk, 0, DM, bk,
                                                   Kb, 0, 0, 1, QDIM, (int)BL, QDIM, DM, 1.f);
    wmma_gemm_kernel<1,1,0><<<g, blk, 0, stream>>>(x, 0, DM, Wv, 0, DM, bv,
                                                   V, 0, 0, 1, QDIM, (int)BL, QDIM, DM, 1.f);
  }

  // 3) pooling convs as GEMMs (body rows) + cls pass-through
  {
    dim3 gp(HD / 128, (BODY + 127) / 128, BATCH * NHEAD);
    pool_gemm_kernel<<<gp, blk, 0, stream>>>(Q, QDIM, NHEAD, WRQ, PQ);
    pool_gemm_kernel<<<gp, blk, 0, stream>>>(Kb, QDIM, NHEAD, WRK, PK);
    pool_gemm_kernel<<<gp, blk, 0, stream>>>(V, QDIM, NHEAD, WRV, PV);
    dim3 gx(HD / 128, (BODY + 127) / 128, BATCH);
    pool_gemm_kernel<<<gx, blk, 0, stream>>>(x, DM, 1, WRX, PX);
  }
  cls_copy_kernel<<<(BATCH * NHEAD * HD + 255) / 256, blk, 0, stream>>>(Q, QDIM, NHEAD, PQ, BATCH * NHEAD);
  cls_copy_kernel<<<(BATCH * NHEAD * HD + 255) / 256, blk, 0, stream>>>(Kb, QDIM, NHEAD, PK, BATCH * NHEAD);
  cls_copy_kernel<<<(BATCH * NHEAD * HD + 255) / 256, blk, 0, stream>>>(V, QDIM, NHEAD, PV, BATCH * NHEAD);
  cls_copy_kernel<<<(BATCH * HD + 255) / 256, blk, 0, stream>>>(x, DM, 1, PX, BATCH);

  // 4) logits = scale * PQ @ PK^T    (batched over 16 (b,n))
  const float scale = 0.044194173824159216f;  // 512^-0.5
  {
    dim3 g((LP + 127) / 128, (LP + 127) / 128, BATCH * NHEAD);
    wmma_gemm_kernel<1,1,0><<<g, blk, 0, stream>>>(PQ, (long long)LP * HD, HD,
                                                   PK, (long long)LP * HD, HD, nullptr,
                                                   LG, (long long)LP * LP, 0, 1, LP,
                                                   LP, LP, HD, scale);
  }
  // 5) logits[1:,1:] += PQ[1:] @ emb^T
  {
    dim3 g((BODY + 127) / 128, (BODY + 127) / 128, BATCH * NHEAD);
    wmma_gemm_kernel<1,1,1><<<g, blk, 0, stream>>>(PQ + HD, (long long)LP * HD, HD,
                                                   EM, 0, HD, nullptr,
                                                   LG + LP + 1, (long long)LP * LP, 0, 1, LP,
                                                   BODY, BODY, HD, 1.f);
  }

  // 6) softmax rows
  softmax_kernel<<<BATCH * NHEAD * LP, blk, 0, stream>>>(LG);

  // 7) attn @ PV -> stacked (B,L2,4096) layout directly
  //    A = attn (lda=1569, unaligned -> scalar staging), B = PV [K,N] row-major
  {
    dim3 g(HD / 128, (LP + 127) / 128, BATCH * NHEAD);
    wmma_gemm_kernel<0,0,0><<<g, blk, 0, stream>>>(LG, (long long)LP * LP, LP,
                                                   PV, (long long)LP * HD, HD, nullptr,
                                                   ST, (long long)LP * QDIM, HD, NHEAD, QDIM,
                                                   LP, HD, LP, 1.f);
  }

  // 8) residual: stacked += (q==0?1:2) * PQ
  {
    long long tot = (long long)BATCH * LP * QDIM;
    residual_kernel<<<(int)((tot + 255) / 256), blk, 0, stream>>>(ST, PQ);
  }

  // 9) y = stacked @ Wd^T + bd
  {
    dim3 g(DM / 128, (BATCH * LP + 127) / 128, 1);
    wmma_gemm_kernel<1,1,0><<<g, blk, 0, stream>>>(ST, 0, QDIM, Wd, 0, QDIM, bd,
                                                   Y, 0, 0, 1, DM,
                                                   BATCH * LP, DM, QDIM, 1.f);
  }

  // 10) out = LN(y + PX) * gamma + beta
  ln_kernel<<<BATCH * LP, blk, 0, stream>>>(Y, PX, gamma, beta, out);
}
